// PointTransformerLayer_6442450944537
// MI455X (gfx1250) — matrix operand order repaired and checked by reading, not compiled, not run on hardware
//
#include <hip/hip_runtime.h>
#include <hip/hip_bf16.h>
#include <math.h>

// Problem constants (match reference)
#define BB    4
#define NPTS  4096
#define DIMF  128
#define POSH  64
#define HID   512          // DIM * ATTN_MULT
#define KNNK  16
#define MTOT  (BB * NPTS)  // 16384 rows
#define PPB   4            // points per attention block
#define MROW  (PPB * KNNK) // 64 activation rows per attention block

// LDS row paddings: dword row-stride % 64 == 4*odd -> conflict-free 16-lane
// ds_load_b128, rows stay 16-byte aligned.
#define PH_LD  (POSH + 8)   // 72 bf16  = 144B  = 36 dw
#define X_LD   (DIMF + 8)   // 136 bf16 = 272B  = 68 dw
#define H_LD   (HID + 8)    // 520 bf16 = 1040B = 260 dw

// Shared-memory layout for attn_kernel (single block-shared arena, 16B aligned)
#define S_IDX_OFF   0                       //  64 * 4      =   256
#define S_REL_OFF   256                     //  64 * 4 * 4  =  1024
#define S_PH_OFF    1280                    //  64*72*2     =  9216
#define S_POS_OFF   10496                   //  64*128*2    = 16384
#define S_X_OFF     26880                   //  64*136*2    = 17408
#define S_VP_OFF    44288                   //  64*128*2    = 16384
#define S_H_OFF     60672                   //  64*520*2    = 66560 (alias: kraw f32 64x128)
#define S_A_OFF     127232                  //  64*128*4    = 32768 (alias: vraw f32 64x128)
#define SMEM_BYTES  160000

typedef __bf16 bf16;
typedef __attribute__((ext_vector_type(16))) __bf16 v16bf;
typedef __attribute__((ext_vector_type(8)))  __bf16 v8bf;
typedef __attribute__((ext_vector_type(8)))  float  v8f;

// ---------------------------------------------------------------------------
// WMMA fragment helpers (wave32, V_WMMA_F32_16X16X32_BF16)
// ---------------------------------------------------------------------------
__device__ __forceinline__ v16bf load_a_bf16(const bf16* A, int lda, int lane) {
  int m  = lane & 15;
  int kh = (lane >> 4) << 3;   // 0 or 8
  const bf16* row = A + m * lda + kh;
  v8bf lo = *(const v8bf*)(row);
  v8bf hi = *(const v8bf*)(row + 16);
  return __builtin_shufflevector(lo, hi, 0, 1, 2, 3, 4, 5, 6, 7,
                                 8, 9, 10, 11, 12, 13, 14, 15);
}

// B fragments pre-swizzled: packed[((nt*kSteps+ks)*32+lane)*16 + j]
__device__ __forceinline__ v16bf load_b_packed(const bf16* P, int kSteps,
                                               int nt, int ks, int lane) {
  const bf16* base = P + ((((size_t)nt * kSteps + ks) * 32 + lane) << 4);
  v8bf lo = *(const v8bf*)(base);
  v8bf hi = *(const v8bf*)(base + 8);
  return __builtin_shufflevector(lo, hi, 0, 1, 2, 3, 4, 5, 6, 7,
                                 8, 9, 10, 11, 12, 13, 14, 15);
}

__device__ __forceinline__ v8f wmma_bf16(v16bf a, v16bf b, v8f c) {
  return __builtin_amdgcn_wmma_f32_16x16x32_bf16(false, a, false, b, (short)0, c,
                                                 false, false);
}

// ---------------------------------------------------------------------------
// Kernel 1a: f32 -> bf16 conversion (activations)
// ---------------------------------------------------------------------------
__global__ __launch_bounds__(256) void cvt_bf16_kernel(const float* __restrict__ src,
                                                       bf16* __restrict__ dst, int n) {
  int i = blockIdx.x * 256 + threadIdx.x;
  if (i < n) dst[i] = (bf16)src[i];
}

// ---------------------------------------------------------------------------
// Kernel 1b: pack f32 weight [Kdim x Ndim] into bf16 B-fragment order
// ---------------------------------------------------------------------------
__global__ __launch_bounds__(256) void pack_b_kernel(const float* __restrict__ W,
                                                     bf16* __restrict__ packed,
                                                     int Kdim, int Ndim) {
  int i = blockIdx.x * 256 + threadIdx.x;
  if (i >= Kdim * Ndim) return;
  int j    = i & 15;
  int lane = (i >> 4) & 31;
  int rest = i >> 9;                 // nt*kSteps + ks
  int kSteps = Kdim >> 5;
  int ks = rest % kSteps;
  int nt = rest / kSteps;
  int krow = ks * 32 + ((j & 8) << 1) + ((lane >> 4) << 3) + (j & 7);
  int ncol = nt * 16 + (lane & 15);
  packed[i] = (bf16)W[(size_t)krow * Ndim + ncol];
}

// ---------------------------------------------------------------------------
// Kernel 2: brute-force KNN (top-16 by squared distance, self included)
// ---------------------------------------------------------------------------
__global__ __launch_bounds__(128) void knn_kernel(const float* __restrict__ xyz,
                                                  int* __restrict__ knn_out) {
  const int b  = blockIdx.x / (NPTS / 128);
  const int qi = (blockIdx.x % (NPTS / 128)) * 128 + threadIdx.x;
  const float* X = xyz + (size_t)b * NPTS * 3;

  const float qx = X[qi * 3 + 0], qy = X[qi * 3 + 1], qz = X[qi * 3 + 2];

  float bestd[KNNK];
  int   besti[KNNK];
#pragma unroll
  for (int s = 0; s < KNNK; ++s) { bestd[s] = 3.4e38f; besti[s] = 0; }

  __shared__ float sc[128 * 3];

  for (int tile = 0; tile < NPTS; tile += 128) {
    __syncthreads();
#pragma unroll
    for (int e = threadIdx.x; e < 128 * 3; e += 128) sc[e] = X[tile * 3 + e];
    __syncthreads();

    for (int c = 0; c < 128; ++c) {
      float dx = qx - sc[c * 3 + 0];
      float dy = qy - sc[c * 3 + 1];
      float dz = qz - sc[c * 3 + 2];
      float d  = dx * dx + dy * dy + dz * dz;
      int   j  = tile + c;
      if (d < bestd[KNNK - 1]) {
#pragma unroll
        for (int s = KNNK - 1; s >= 1; --s) {
          if (d < bestd[s - 1])  { bestd[s] = bestd[s - 1]; besti[s] = besti[s - 1]; }
          else if (d < bestd[s]) { bestd[s] = d;            besti[s] = j; }
        }
        if (d < bestd[0]) { bestd[0] = d; besti[0] = j; }
      }
    }
  }
#pragma unroll
  for (int s = 0; s < KNNK; ++s)
    knn_out[((size_t)b * NPTS + qi) * KNNK + s] = besti[s];
}

// ---------------------------------------------------------------------------
// Kernel 3: q/k/v projections  [MTOT,128] x [128,128] (x3), bf16 WMMA, f32 out
// ---------------------------------------------------------------------------
__global__ __launch_bounds__(128) void qkv_kernel(const bf16* __restrict__ feat,
                                                  const bf16* __restrict__ wq,
                                                  const bf16* __restrict__ wk,
                                                  const bf16* __restrict__ wv,
                                                  float* __restrict__ qo,
                                                  float* __restrict__ ko,
                                                  float* __restrict__ vo) {
  const int lane = threadIdx.x & 31;
  const int wave = blockIdx.x * 4 + (threadIdx.x >> 5);
  const int MTILES = MTOT / 16;  // 1024
  const int NTILES = DIMF / 16;  // 8
  const int which  = wave / (MTILES * NTILES);
  const int rem    = wave % (MTILES * NTILES);
  const int mt = rem / NTILES, nt = rem % NTILES;

  const bf16* W = (which == 0) ? wq : (which == 1) ? wk : wv;
  float*      O = (which == 0) ? qo : (which == 1) ? ko : vo;

  const bf16* Ab = feat + (size_t)(mt * 16) * DIMF;
  v8f c = {};
#pragma unroll
  for (int ks = 0; ks < DIMF / 32; ++ks) {
    v16bf a = load_a_bf16(Ab + ks * 32, DIMF, lane);
    v16bf b = load_b_packed(W, DIMF / 32, nt, ks, lane);
    c = wmma_bf16(a, b, c);
  }
  const int col = lane & 15;
#pragma unroll
  for (int r = 0; r < 8; ++r) {
    int m = mt * 16 + ((lane >> 4) * 8 + r);
    O[(size_t)m * DIMF + nt * 16 + col] = c[r];
  }
}

// ---------------------------------------------------------------------------
// Kernel 4: fused attention, PPB=4 points (M=64 rows) per 256-thread block.
// B fragments register-cached across the 4 M-tiles; k/v gathers staged to LDS
// with async DMA overlapped with the pos-MLP.
// ---------------------------------------------------------------------------
__global__ __launch_bounds__(256) void attn_kernel(
    const float* __restrict__ xyz,
    const float* __restrict__ qm, const float* __restrict__ km,
    const float* __restrict__ vm, const int* __restrict__ knn,
    const float* __restrict__ pW1, const float* __restrict__ pb1,
    const bf16* __restrict__ pw2p, const float* __restrict__ pb2,
    const bf16* __restrict__ aw1p, const float* __restrict__ ab1,
    const bf16* __restrict__ aw2p, const float* __restrict__ ab2,
    bf16* __restrict__ agg) {
  const int p0 = blockIdx.x * PPB;   // global point row base (within MTOT)
  const int b  = p0 / NPTS;          // PPB divides NPTS -> no batch straddle
  const int t  = threadIdx.x;        // 0..255
  const int lane = t & 31;
  const int wid  = t >> 5;           // 0..7

  __shared__ __attribute__((aligned(16))) char smem[SMEM_BYTES];
  int*   s_idx = (int*)(smem + S_IDX_OFF);          // [MROW]
  float* s_rel = (float*)(smem + S_REL_OFF);        // [MROW][4]
  bf16*  s_ph  = (bf16*)(smem + S_PH_OFF);          // [MROW][PH_LD]
  bf16*  s_pos = (bf16*)(smem + S_POS_OFF);         // [MROW][DIMF]
  bf16*  s_X   = (bf16*)(smem + S_X_OFF);           // [MROW][X_LD]
  bf16*  s_vp  = (bf16*)(smem + S_VP_OFF);          // [MROW][DIMF]
  bf16*  s_H   = (bf16*)(smem + S_H_OFF);           // [MROW][H_LD]
  float* s_A   = (float*)(smem + S_A_OFF);          // [MROW][DIMF]
  float* kraw  = (float*)(smem + S_H_OFF);          // alias (pre-L1)
  float* vraw  = (float*)(smem + S_A_OFF);          // alias (pre-L2)

  // ---- neighbor indices + relative positions -----------------------------
  if (t < MROW) {
    int pp = t >> 4, kk = t & 15;
    int pt = p0 + pp;
    int j  = knn[(size_t)pt * KNNK + kk];
    s_idx[t] = j;
    const float* xq = xyz + (size_t)pt * 3;
    const float* xc = xyz + ((size_t)b * NPTS + j) * 3;
    s_rel[t * 4 + 0] = xq[0] - xc[0];
    s_rel[t * 4 + 1] = xq[1] - xc[1];
    s_rel[t * 4 + 2] = xq[2] - xc[2];
  }
  __syncthreads();

  // ---- async DMA: gather k/v rows (512B each) into LDS staging -----------
  // overlaps with both pos-MLP layers below
  for (int job = wid; job < 2 * MROW; job += 8) {
    int isv = job >> 6;             // 0: k rows, 1: v rows
    int r   = job & (MROW - 1);
    int nb  = s_idx[r];
    const float* src = (isv ? vm : km) + ((size_t)b * NPTS + nb) * DIMF;
    unsigned long long ga = (unsigned long long)(uintptr_t)src +
                            (unsigned long long)(lane * 16);
    unsigned int lo = (isv ? S_A_OFF : S_H_OFF) + r * (DIMF * 4) + lane * 16;
    asm volatile("global_load_async_to_lds_b128 %0, %1, off"
                 :: "v"(lo), "v"(ga) : "memory");
  }

  // ---- pos-MLP layer 1 (tiny, VALU): 64x64 outputs, 16 per thread --------
#pragma unroll
  for (int o = t * 16; o < t * 16 + 16; ++o) {
    int kk = o >> 6, h = o & 63;
    float acc = pb1[h]
              + s_rel[kk * 4 + 0] * pW1[0 * POSH + h]
              + s_rel[kk * 4 + 1] * pW1[1 * POSH + h]
              + s_rel[kk * 4 + 2] * pW1[2 * POSH + h];
    s_ph[kk * PH_LD + h] = (bf16)fmaxf(acc, 0.0f);
  }
  __syncthreads();

  // ---- pos-MLP layer 2 WMMA: [64,64]@[64,128]; wave w -> n-tile w --------
  {
    const int nt = wid;
    v16bf b0 = load_b_packed(pw2p, POSH / 32, nt, 0, lane);
    v16bf b1 = load_b_packed(pw2p, POSH / 32, nt, 1, lane);
    int col = lane & 15;
#pragma unroll
    for (int mt = 0; mt < PPB; ++mt) {
      v8f c = {};
      c = wmma_bf16(load_a_bf16(s_ph + (mt * 16) * PH_LD, PH_LD, lane), b0, c);
      c = wmma_bf16(load_a_bf16(s_ph + (mt * 16) * PH_LD + 32, PH_LD, lane), b1, c);
#pragma unroll
      for (int r = 0; r < 8; ++r) {
        int ml = mt * 16 + (lane >> 4) * 8 + r;
        s_pos[ml * DIMF + nt * 16 + col] = (bf16)(c[r] + pb2[nt * 16 + col]);
      }
    }
  }
  // drain this wave's async DMA, then block-wide visibility
  asm volatile("s_wait_asynccnt 0x0" ::: "memory");
  __syncthreads();

  // ---- build X = q - k + pos (bf16) and vp = v + pos (bf16) --------------
  {
    const float4* qm4 = (const float4*)qm;
    const float4* kr4 = (const float4*)kraw;
    const float4* vr4 = (const float4*)vraw;
    const int ROW4 = DIMF / 4;  // 32
#pragma unroll
    for (int e = t; e < MROW * ROW4; e += 256) {
      int r = e >> 5, f4 = e & 31;
      int pp = r >> 4;
      float4 kv = kr4[r * ROW4 + f4];
      float4 vv = vr4[r * ROW4 + f4];
      float4 qv = qm4[(size_t)(p0 + pp) * ROW4 + f4];
      int f = f4 * 4;
      float pz0 = (float)s_pos[r * DIMF + f + 0];
      float pz1 = (float)s_pos[r * DIMF + f + 1];
      float pz2 = (float)s_pos[r * DIMF + f + 2];
      float pz3 = (float)s_pos[r * DIMF + f + 3];
      s_X[r * X_LD + f + 0] = (bf16)(qv.x - kv.x + pz0);
      s_X[r * X_LD + f + 1] = (bf16)(qv.y - kv.y + pz1);
      s_X[r * X_LD + f + 2] = (bf16)(qv.z - kv.z + pz2);
      s_X[r * X_LD + f + 3] = (bf16)(qv.w - kv.w + pz3);
      s_vp[r * DIMF + f + 0] = (bf16)(vv.x + pz0);
      s_vp[r * DIMF + f + 1] = (bf16)(vv.y + pz1);
      s_vp[r * DIMF + f + 2] = (bf16)(vv.z + pz2);
      s_vp[r * DIMF + f + 3] = (bf16)(vv.w + pz3);
    }
  }
  __syncthreads();

  // ---- attn-MLP layer 1: [64,128]@[128,512], relu ------------------------
  // wave w handles n-tiles {w, w+8, w+16, w+24}; B frags cached over 4 M-tiles
#pragma unroll
  for (int g = 0; g < 4; ++g) {
    const int nt = wid + g * 8;
    v16bf b0 = load_b_packed(aw1p, DIMF / 32, nt, 0, lane);
    v16bf b1 = load_b_packed(aw1p, DIMF / 32, nt, 1, lane);
    v16bf b2 = load_b_packed(aw1p, DIMF / 32, nt, 2, lane);
    v16bf b3 = load_b_packed(aw1p, DIMF / 32, nt, 3, lane);
    int col = lane & 15;
#pragma unroll
    for (int mt = 0; mt < PPB; ++mt) {
      const bf16* Xa = s_X + (mt * 16) * X_LD;
      v8f c = {};
      c = wmma_bf16(load_a_bf16(Xa + 0,  X_LD, lane), b0, c);
      c = wmma_bf16(load_a_bf16(Xa + 32, X_LD, lane), b1, c);
      c = wmma_bf16(load_a_bf16(Xa + 64, X_LD, lane), b2, c);
      c = wmma_bf16(load_a_bf16(Xa + 96, X_LD, lane), b3, c);
#pragma unroll
      for (int r = 0; r < 8; ++r) {
        int ml = mt * 16 + (lane >> 4) * 8 + r;
        s_H[ml * H_LD + nt * 16 + col] =
            (bf16)fmaxf(c[r] + ab1[nt * 16 + col], 0.0f);
      }
    }
  }
  __syncthreads();

  // ---- attn-MLP layer 2: [64,512]@[512,128] ------------------------------
  // wave w -> n-tile w; 4 per-M-tile accumulators; B frag loaded once per ks
  {
    const int nt = wid;
    v8f c0 = {}, c1 = {}, c2 = {}, c3 = {};
#pragma unroll
    for (int ks = 0; ks < HID / 32; ++ks) {
      v16bf bfr = load_b_packed(aw2p, HID / 32, nt, ks, lane);
      c0 = wmma_bf16(load_a_bf16(s_H + 0 * 16 * H_LD + ks * 32, H_LD, lane), bfr, c0);
      c1 = wmma_bf16(load_a_bf16(s_H + 1 * 16 * H_LD + ks * 32, H_LD, lane), bfr, c1);
      c2 = wmma_bf16(load_a_bf16(s_H + 2 * 16 * H_LD + ks * 32, H_LD, lane), bfr, c2);
      c3 = wmma_bf16(load_a_bf16(s_H + 3 * 16 * H_LD + ks * 32, H_LD, lane), bfr, c3);
    }
    int col = lane & 15;
    float bias = ab2[nt * 16 + col];
#pragma unroll
    for (int r = 0; r < 8; ++r) {
      int ml = (lane >> 4) * 8 + r;
      s_A[(0 * 16 + ml) * DIMF + nt * 16 + col] = c0[r] + bias;
      s_A[(1 * 16 + ml) * DIMF + nt * 16 + col] = c1[r] + bias;
      s_A[(2 * 16 + ml) * DIMF + nt * 16 + col] = c2[r] + bias;
      s_A[(3 * 16 + ml) * DIMF + nt * 16 + col] = c3[r] + bias;
    }
  }
  __syncthreads();

  // ---- softmax over 16 neighbors per (point, channel); aggregate ---------
  {
    const float scale = 0.088388347648318447f;  // 1/sqrt(128)
#pragma unroll
    for (int job = t; job < PPB * DIMF; job += 256) {
      int pp = job >> 7, f = job & 127;
      const float* Ac = s_A + (pp * 16) * DIMF + f;
      const bf16*  Vc = s_vp + (pp * 16) * DIMF + f;
      float mx = -3.4e38f;
#pragma unroll
      for (int kk = 0; kk < KNNK; ++kk) mx = fmaxf(mx, Ac[kk * DIMF] * scale);
      float sum = 0.0f, acc = 0.0f;
#pragma unroll
      for (int kk = 0; kk < KNNK; ++kk) {
        float ex = __expf(Ac[kk * DIMF] * scale - mx);
        sum += ex;
        acc += ex * (float)Vc[kk * DIMF];
      }
      agg[(size_t)(p0 + pp) * DIMF + f] = (bf16)(acc / sum);
    }
  }
}

// ---------------------------------------------------------------------------
// Kernel 5: out = agg @ fW + fb + feature   (f32 out)
// ---------------------------------------------------------------------------
__global__ __launch_bounds__(128) void final_kernel(const bf16* __restrict__ agg,
                                                    const bf16* __restrict__ fwp,
                                                    const float* __restrict__ fb,
                                                    const float* __restrict__ feat,
                                                    float* __restrict__ out) {
  const int lane = threadIdx.x & 31;
  const int wave = blockIdx.x * 4 + (threadIdx.x >> 5);
  const int NTILES = DIMF / 16;  // 8
  const int mt = wave / NTILES, nt = wave % NTILES;

  const bf16* Ab = agg + (size_t)(mt * 16) * DIMF;
  v8f c = {};
#pragma unroll
  for (int ks = 0; ks < DIMF / 32; ++ks) {
    v16bf a = load_a_bf16(Ab + ks * 32, DIMF, lane);
    v16bf b = load_b_packed(fwp, DIMF / 32, nt, ks, lane);
    c = wmma_bf16(a, b, c);
  }
  const int col = lane & 15;
#pragma unroll
  for (int r = 0; r < 8; ++r) {
    int m = mt * 16 + ((lane >> 4) * 8 + r);
    int f = nt * 16 + col;
    out[(size_t)m * DIMF + f] = c[r] + fb[f] + feat[(size_t)m * DIMF + f];
  }
}

// ---------------------------------------------------------------------------
// Host launcher
// ---------------------------------------------------------------------------
extern "C" void kernel_launch(void* const* d_in, const int* in_sizes, int n_in,
                              void* d_out, int out_size, void* d_ws, size_t ws_size,
                              hipStream_t stream) {
  const float* xyz  = (const float*)d_in[0];
  const float* feat = (const float*)d_in[1];
  const float* Wq   = (const float*)d_in[2];
  const float* Wk   = (const float*)d_in[3];
  const float* Wv   = (const float*)d_in[4];
  const float* pW1  = (const float*)d_in[5];
  const float* pb1  = (const float*)d_in[6];
  const float* pW2  = (const float*)d_in[7];
  const float* pb2  = (const float*)d_in[8];
  const float* aW1  = (const float*)d_in[9];
  const float* ab1  = (const float*)d_in[10];
  const float* aW2  = (const float*)d_in[11];
  const float* ab2  = (const float*)d_in[12];
  const float* fW   = (const float*)d_in[13];
  const float* fb   = (const float*)d_in[14];
  float* out = (float*)d_out;

  char* ws = (char*)d_ws;
  size_t off = 0;
  auto alloc = [&](size_t bytes) -> void* {
    void* pp = (void*)(ws + off);
    off = (off + bytes + 255) & ~(size_t)255;
    return pp;
  };

  bf16* feat_bf = (bf16*)alloc((size_t)MTOT * DIMF * 2);
  bf16* wq_pk   = (bf16*)alloc((size_t)DIMF * DIMF * 2);
  bf16* wk_pk   = (bf16*)alloc((size_t)DIMF * DIMF * 2);
  bf16* wv_pk   = (bf16*)alloc((size_t)DIMF * DIMF * 2);
  bf16* pw2_pk  = (bf16*)alloc((size_t)POSH * DIMF * 2);
  bf16* aw1_pk  = (bf16*)alloc((size_t)DIMF * HID * 2);
  bf16* aw2_pk  = (bf16*)alloc((size_t)HID * DIMF * 2);
  bf16* fw_pk   = (bf16*)alloc((size_t)DIMF * DIMF * 2);
  float* qf     = (float*)alloc((size_t)MTOT * DIMF * 4);
  float* kf     = (float*)alloc((size_t)MTOT * DIMF * 4);
  float* vf     = (float*)alloc((size_t)MTOT * DIMF * 4);
  int*   knn    = (int*)alloc((size_t)MTOT * KNNK * 4);
  bf16*  agg    = (bf16*)alloc((size_t)MTOT * DIMF * 2);

  cvt_bf16_kernel<<<(MTOT * DIMF + 255) / 256, 256, 0, stream>>>(feat, feat_bf,
                                                                 MTOT * DIMF);
  auto pack = [&](const float* s, bf16* d, int Kd, int Nd) {
    pack_b_kernel<<<(Kd * Nd + 255) / 256, 256, 0, stream>>>(s, d, Kd, Nd);
  };
  pack(Wq, wq_pk, DIMF, DIMF);
  pack(Wk, wk_pk, DIMF, DIMF);
  pack(Wv, wv_pk, DIMF, DIMF);
  pack(pW2, pw2_pk, POSH, DIMF);
  pack(aW1, aw1_pk, DIMF, HID);
  pack(aW2, aw2_pk, HID, DIMF);
  pack(fW, fw_pk, DIMF, DIMF);

  knn_kernel<<<BB * (NPTS / 128), 128, 0, stream>>>(xyz, knn);

  qkv_kernel<<<3 * (MTOT / 16) * (DIMF / 16) / 4, 128, 0, stream>>>(
      feat_bf, wq_pk, wk_pk, wv_pk, qf, kf, vf);

  attn_kernel<<<MTOT / PPB, 256, 0, stream>>>(xyz, qf, kf, vf, knn, pW1, pb1,
                                              pw2_pk, pb2, aw1_pk, ab1, aw2_pk,
                                              ab2, agg);

  final_kernel<<<(MTOT / 16) * (DIMF / 16) / 4, 128, 0, stream>>>(agg, fw_pk, fb,
                                                                  feat, out);
  (void)in_sizes; (void)n_in; (void)out_size; (void)ws_size;
}
